// SplitMessagePass_11965778886799
// MI455X (gfx1250) — compile-verified, matrix operand order checked by reading, not run on hardware
//
#include <hip/hip_runtime.h>
#include <hip/hip_bf16.h>

#define N_NODES 100000
#define N_EDGES 1600000
#define IN_F    64
#define OUT_F   64
#define OUT_COLS 192   // 3 * OUT_F concatenated
#define ROW_BLOCKS (N_NODES / 16)   // 6250, exact
#define N_FRAGS (3 * 4 * 2)         // mats * ntiles * ksteps = 24

typedef __attribute__((ext_vector_type(16))) _Float16 v16h;
typedef __attribute__((ext_vector_type(8)))  float    v8f;

// ---------------------------------------------------------------------------
// Kernel 1: edge scatter with hardware f32 atomics.
// 16 lanes per edge, each lane handles 4 contiguous floats (one float4).
// label in {0,1}: pos scale = label+1 (always added), neg scale = label
// (added only when label==1, and then the scale is exactly 1).
// ---------------------------------------------------------------------------
__global__ void smp_scatter_kernel(const float* __restrict__ x,
                                   const int*   __restrict__ label,
                                   const int*   __restrict__ src,
                                   const int*   __restrict__ dst,
                                   float*       __restrict__ pos,
                                   float*       __restrict__ neg) {
    int t    = blockIdx.x * blockDim.x + threadIdx.x;
    int e    = t >> 4;
    int l16  = t & 15;
    if (e >= N_EDGES) return;

    int s   = src[e];
    int d   = dst[e];
    int lbl = label[s];

    const float4 v = *reinterpret_cast<const float4*>(x + (size_t)s * IN_F + l16 * 4);
    float sp = (float)(lbl + 1);

    float* pp = pos + (size_t)d * IN_F + l16 * 4;
    unsafeAtomicAdd(pp + 0, v.x * sp);
    unsafeAtomicAdd(pp + 1, v.y * sp);
    unsafeAtomicAdd(pp + 2, v.z * sp);
    unsafeAtomicAdd(pp + 3, v.w * sp);

    if (lbl) {  // neg scale == 1 exactly when lbl==1, else 0 (skip)
        float* pn = neg + (size_t)d * IN_F + l16 * 4;
        unsafeAtomicAdd(pn + 0, v.x);
        unsafeAtomicAdd(pn + 1, v.y);
        unsafeAtomicAdd(pn + 2, v.z);
        unsafeAtomicAdd(pn + 3, v.w);
    }
}

// ---------------------------------------------------------------------------
// Kernel 2: pre-pack the three weight matrices into WMMA B-fragment layout.
// Fragment f = ((mat*4 + t)*2 + ks); per lane a v16h (32 B):
//   n  = t*16 + (lane&15), hi = lane>>4
//   b[i]   = W[(ks*32 + hi*8 + i)      * 64 + n]   (i < 8)
//   b[i+8] = W[(ks*32 + hi*8 + i + 16) * 64 + n]
// Total 24 frags * 32 lanes * 32 B = 24 KB, L0/L2-resident for all GEMM waves.
// ---------------------------------------------------------------------------
__global__ void smp_pack_b_kernel(const float* __restrict__ Wself,
                                  const float* __restrict__ Wpos,
                                  const float* __restrict__ Wneg,
                                  v16h*        __restrict__ bfrag) {
    int u = blockIdx.x * blockDim.x + threadIdx.x;
    if (u >= N_FRAGS * 32) return;
    int lane = u & 31;
    int f    = u >> 5;
    int ks   = f & 1;
    int t    = (f >> 1) & 3;
    int mat  = f >> 3;

    const float* W = (mat == 0) ? Wself : ((mat == 1) ? Wpos : Wneg);
    int n  = t * 16 + (lane & 15);
    int hi = lane >> 4;

    v16h b;
#pragma unroll
    for (int i = 0; i < 8; ++i) {
        int k = ks * 32 + hi * 8 + i;
        b[i]     = (_Float16)W[(size_t)k        * OUT_F + n];
        b[i + 8] = (_Float16)W[(size_t)(k + 16) * OUT_F + n];
    }
    bfrag[u] = b;
}

// ---------------------------------------------------------------------------
// Kernel 3: WMMA GEMM + ReLU.
// One wave per (16-row block, matrix). mat: 0 = W_self (A = x + (1+ep)*pos +
// (1+en)*neg), 1 = W_pos (A = pos), 2 = W_neg (A = neg).
// A fragment layout (16-bit A 16x32, wave32): lane L, m = L&15, hi = L>>4:
//   element i<8  -> K = ks*32 + hi*8 + i
//   element i>=8 -> K = ks*32 + 16 + hi*8 + (i-8)
// B fragments come pre-packed from smp_pack_b_kernel (one 32B load per lane).
// C/D v8f: VGPR r -> row r + hi*8, col = L&15.
// ---------------------------------------------------------------------------
__global__ void smp_gemm_kernel(const float* __restrict__ x,
                                const float* __restrict__ pos,
                                const float* __restrict__ neg,
                                const v16h*  __restrict__ bfrag,
                                const float* __restrict__ eps_pos,
                                const float* __restrict__ eps_neg,
                                float*       __restrict__ out) {
    const int lane = threadIdx.x & 31;
    const int gw   = blockIdx.x * (blockDim.x >> 5) + (threadIdx.x >> 5);
    const int mat  = gw % 3;
    const int rb   = gw / 3;
    if (rb >= ROW_BLOCKS) return;          // uniform per wave

    const int m   = lane & 15;
    const int hi  = lane >> 4;
    const int row = rb * 16 + m;

    const float ep = 1.0f + eps_pos[0];
    const float en = 1.0f + eps_neg[0];

    // ---- load this lane's 32 A elements: 4 chunks of 8 floats at
    //      K offsets hi*8 + {0, 16, 32, 48} of row `row` ----
    float areg[32];
    const size_t rbase = (size_t)row * IN_F;
#pragma unroll
    for (int c = 0; c < 4; ++c) {
        const size_t off = rbase + c * 16 + hi * 8;
        if (mat == 0) {
            float4 vx0 = *reinterpret_cast<const float4*>(x   + off);
            float4 vx1 = *reinterpret_cast<const float4*>(x   + off + 4);
            float4 vp0 = *reinterpret_cast<const float4*>(pos + off);
            float4 vp1 = *reinterpret_cast<const float4*>(pos + off + 4);
            float4 vn0 = *reinterpret_cast<const float4*>(neg + off);
            float4 vn1 = *reinterpret_cast<const float4*>(neg + off + 4);
            areg[c*8+0] = vx0.x + ep*vp0.x + en*vn0.x;
            areg[c*8+1] = vx0.y + ep*vp0.y + en*vn0.y;
            areg[c*8+2] = vx0.z + ep*vp0.z + en*vn0.z;
            areg[c*8+3] = vx0.w + ep*vp0.w + en*vn0.w;
            areg[c*8+4] = vx1.x + ep*vp1.x + en*vn1.x;
            areg[c*8+5] = vx1.y + ep*vp1.y + en*vn1.y;
            areg[c*8+6] = vx1.z + ep*vp1.z + en*vn1.z;
            areg[c*8+7] = vx1.w + ep*vp1.w + en*vn1.w;
        } else {
            const float* a = (mat == 1) ? pos : neg;
            float4 v0 = *reinterpret_cast<const float4*>(a + off);
            float4 v1 = *reinterpret_cast<const float4*>(a + off + 4);
            areg[c*8+0] = v0.x; areg[c*8+1] = v0.y;
            areg[c*8+2] = v0.z; areg[c*8+3] = v0.w;
            areg[c*8+4] = v1.x; areg[c*8+5] = v1.y;
            areg[c*8+6] = v1.z; areg[c*8+7] = v1.w;
        }
    }

    v8f acc[4];
#pragma unroll
    for (int t = 0; t < 4; ++t)
#pragma unroll
        for (int r = 0; r < 8; ++r) acc[t][r] = 0.0f;

#pragma unroll
    for (int ks = 0; ks < 2; ++ks) {
        // A fragment for this K step: exactly areg[ks*16 .. ks*16+15]
        v16h a;
#pragma unroll
        for (int i = 0; i < 16; ++i) a[i] = (_Float16)areg[ks * 16 + i];

#pragma unroll
        for (int t = 0; t < 4; ++t) {
            // pre-packed B fragment: one 32-byte per-lane load (2x b128)
            v16h b = bfrag[(size_t)(((mat * 4 + t) * 2 + ks) * 32 + lane)];
            acc[t] = __builtin_amdgcn_wmma_f32_16x16x32_f16(
                false, a, false, b, (short)0, acc[t], false, false);
        }
    }

    // ---- ReLU + store. VGPR r of acc -> row (rb*16 + r + hi*8), col t*16+m.
#pragma unroll
    for (int t = 0; t < 4; ++t) {
        const int col = mat * OUT_F + t * 16 + m;
#pragma unroll
        for (int r = 0; r < 8; ++r) {
            const int orow = rb * 16 + r + hi * 8;
            float v = acc[t][r];
            out[(size_t)orow * OUT_COLS + col] = v > 0.0f ? v : 0.0f;
        }
    }
}

// ---------------------------------------------------------------------------
extern "C" void kernel_launch(void* const* d_in, const int* in_sizes, int n_in,
                              void* d_out, int out_size, void* d_ws, size_t ws_size,
                              hipStream_t stream) {
    const float* x     = (const float*)d_in[0];
    const int*   label = (const int*)  d_in[1];
    const int*   src   = (const int*)  d_in[2];
    const int*   dst   = (const int*)  d_in[3];
    const float* Wself = (const float*)d_in[4];
    const float* Wpos  = (const float*)d_in[5];
    const float* Wneg  = (const float*)d_in[6];
    const float* eps_p = (const float*)d_in[7];
    const float* eps_n = (const float*)d_in[8];
    float*       out   = (float*)d_out;

    float* pos   = (float*)d_ws;
    float* neg   = pos + (size_t)N_NODES * IN_F;
    v16h*  bfrag = (v16h*)(neg + (size_t)N_NODES * IN_F);  // 32B-aligned offset

    // zero the two accumulators (graph-capture-legal)
    hipMemsetAsync(d_ws, 0, 2 * sizeof(float) * (size_t)N_NODES * IN_F, stream);

    // pack B fragments: 24 frags * 32 lanes = 768 threads
    smp_pack_b_kernel<<<3, 256, 0, stream>>>(Wself, Wpos, Wneg, bfrag);

    // scatter: 16 threads per edge
    {
        const long total = (long)N_EDGES * 16;
        const int  blk   = 256;
        const int  grid  = (int)((total + blk - 1) / blk);   // 100000
        smp_scatter_kernel<<<grid, blk, 0, stream>>>(x, label, src, dst, pos, neg);
    }

    // WMMA GEMM + ReLU: 1 wave per (row block, matrix); 8 waves per block
    {
        const int waves = ROW_BLOCKS * 3;                    // 18750
        const int blk   = 256;                               // 8 waves
        const int grid  = (waves + 7) / 8;                   // 2344
        smp_gemm_kernel<<<grid, blk, 0, stream>>>(x, pos, neg, bfrag,
                                                  eps_p, eps_n, out);
    }
}
